// Attention_29489245454438
// MI455X (gfx1250) — compile-verified, matrix operand order checked by reading, not run on hardware
//
#include <hip/hip_runtime.h>

typedef __bf16 v16bf __attribute__((ext_vector_type(16)));
typedef float  v8f   __attribute__((ext_vector_type(8)));
typedef int    v8i   __attribute__((ext_vector_type(8)));
typedef int    v4i   __attribute__((ext_vector_type(4)));
typedef unsigned int v4u __attribute__((ext_vector_type(4)));
typedef unsigned short u16;
typedef unsigned int   u32;

#define DEVINL __device__ __forceinline__

// Problem constants: B=8, N=1024, C=1024, H=16, HD=64
static constexpr float ATTN_SCALE = 0.125f; // 64^-0.5

DEVINL u16 f32_bf16(float f) {
  u32 u = __builtin_bit_cast(u32, f);
  u = u + 0x7FFFu + ((u >> 16) & 1u);   // round-to-nearest-even
  return (u16)(u >> 16);
}

DEVINL v8f vzero() {
  v8f z;
#pragma unroll
  for (int i = 0; i < 8; ++i) z[i] = 0.0f;
  return z;
}

// 16-bit A/B fragment (16 rows x K=32) per ISA 7.12.2: per lane the 16
// bf16 elements are two contiguous 16-byte runs at u16 offsets
// kg*8 + [0,8) and 16 + kg*8 + [0,8), where kg = lane>>4.
DEVINL v16bf load_frag2(const u16* rowbase, int kg) {
  v4i lo = *(const v4i*)(rowbase + kg * 8);
  v4i hi = *(const v4i*)(rowbase + 16 + kg * 8);
  v8i f = __builtin_shufflevector(lo, hi, 0, 1, 2, 3, 4, 5, 6, 7);
  return __builtin_bit_cast(v16bf, f);
}

DEVINL v8f wmma_bf16(v16bf a, v16bf b, v8f c) {
  return __builtin_amdgcn_wmma_f32_16x16x32_bf16(
      false, a, false, b, (short)0, c, false, false);
}

DEVINL float hmax16(float x) {
#pragma unroll
  for (int m = 1; m < 16; m <<= 1) x = fmaxf(x, __shfl_xor(x, m, 32));
  return x;
}
DEVINL float hsum16(float x) {
#pragma unroll
  for (int m = 1; m < 16; m <<= 1) x += __shfl_xor(x, m, 32);
  return x;
}

// ---- Async global->LDS (per-lane 16B), tracked by ASYNCcnt ----
DEVINL void async_ld_b128(const u16* lds_ptr, const u16* gptr) {
  u32 lds_off = (u32)(size_t)lds_ptr;   // low 32 bits of flat ptr == LDS addr
  unsigned long long ga = (unsigned long long)(size_t)gptr;
  asm volatile("global_load_async_to_lds_b128 %0, %1, off"
               :: "v"(lds_off), "v"(ga) : "memory");
}
DEVINL void wait_async0() {
  asm volatile("s_wait_asynccnt 0" ::: "memory");
}

// ---- Tensor Data Mover: 2-D tile global->LDS, tracked by TENSORcnt ----
// Builds a D# per CDNA5 ISA and issues
// TENSOR_LOAD_TO_LDS (6-arg builtin on this toolchain). tile_d0 = elements
// per row, tile_d1 = rows, stride = row stride in elements (data_size=2B).
// LDS padding: pad_amount+1 DWORDs appended every 2<<pad_interval DWORDs.
DEVINL void tdm_load_2d(u32 lds_addr, const u16* gptr,
                        u32 tile_d0, u32 tile_d1, u32 stride,
                        u32 pad_interval, u32 pad_amount) {
  unsigned long long ga = (unsigned long long)(size_t)gptr;
  const u32 tensor_d0 = stride;        // safe dim0 bound (tile fits)
  const u32 tensor_d1 = 0x100000u;     // large dim1 bound

  v4u g0;
  g0.x = 1u;                                   // count=1, user-mode
  g0.y = lds_addr;                             // [63:32] lds byte addr
  g0.z = (u32)ga;                              // [95:64] gaddr lo
  g0.w = (u32)((ga >> 32) & 0x01FFFFFFull)     // [120:96] gaddr hi
         | (2u << 30);                         // [127:126] type=2 (image)

  v8i g1;
  g1[0] = (int)((1u << 16)                     // data_size = 2 bytes
                | (1u << 20)                   // pad_enable
                | (pad_interval << 22)
                | (pad_amount << 25));
  g1[1] = (int)((tensor_d0 & 0xFFFFu) << 16);          // dim0[15:0]
  g1[2] = (int)(((tensor_d0 >> 16) & 0xFFFFu)          // dim0[31:16]
                | ((tensor_d1 & 0xFFFFu) << 16));      // dim1[15:0]
  g1[3] = (int)(((tensor_d1 >> 16) & 0xFFFFu)          // dim1[31:16]
                | (tile_d0 << 16));                    // tile_dim0
  g1[4] = (int)(tile_d1 & 0xFFFFu);                    // tile_dim1 (dim2=0)
  g1[5] = (int)stride;                                 // dim0_stride lo32
  g1[6] = 0;                                           // stride hi / dim1_stride
  g1[7] = 0;

  v4i g2 = {0, 0, 0, 0};
  v4i g3 = {0, 0, 0, 0};
  v8i g4 = {0, 0, 0, 0, 0, 0, 0, 0};
  __builtin_amdgcn_tensor_load_to_lds(g0, g1, g2, g3, g4, 0);
}

// ---------------- Kernel 1: f32 -> bf16 conversion (x8 vectorized) ----
__global__ void cvt_f32_bf16_v8(const float* __restrict__ in,
                                u16* __restrict__ out, int n8) {
  int i = blockIdx.x * blockDim.x + threadIdx.x;
  int stride = gridDim.x * blockDim.x;
  for (; i < n8; i += stride) {
    float4 a = ((const float4*)in)[2 * i];
    float4 b = ((const float4*)in)[2 * i + 1];
    v4i p;
    p[0] = (int)((u32)f32_bf16(a.x) | ((u32)f32_bf16(a.y) << 16));
    p[1] = (int)((u32)f32_bf16(a.z) | ((u32)f32_bf16(a.w) << 16));
    p[2] = (int)((u32)f32_bf16(b.x) | ((u32)f32_bf16(b.y) << 16));
    p[3] = (int)((u32)f32_bf16(b.z) | ((u32)f32_bf16(b.w) << 16));
    ((v4i*)out)[i] = p;
  }
}

// ---------------- Kernel 2: QKV projection GEMM -------------------
// xb: [8192,1024] bf16, wb: [3072,1024] bf16, qkv[m,n]=sum_k xb[m,k]*wb[n,k]
// Block = 8 waves: 128 M x 64 N tile. K step 64; A(128x64) and B(64x64)
// tiles double-buffered in LDS, moved by the TDM (one TENSOR_LOAD_TO_LDS
// per tile, issued by wave 0; descriptor pads rows 32dw+4dw -> stride 72).
__global__ __launch_bounds__(256) void qkv_gemm(
    const u16* __restrict__ xb, const u16* __restrict__ wb,
    u16* __restrict__ qb, u16* __restrict__ kb, u16* __restrict__ vtb) {
  static constexpr int SA = 72;   // padded row stride (u16)
  static constexpr int SB = 72;
  __shared__ __align__(16) u16 sA[2][128 * SA];
  __shared__ __align__(16) u16 sB[2][64 * SB];

  const int tid  = threadIdx.x;
  const int lane = tid & 31;
  const int wid  = tid >> 5;
  const int r15  = lane & 15;
  const int kg   = lane >> 4;

  const int mblk  = blockIdx.x & 63;   // 8192/128
  const int nblk  = blockIdx.x >> 6;   // 3072/64
  const int m0blk = mblk * 128;
  const int n0    = nblk * 64;

  v8f acc[4];
#pragma unroll
  for (int t = 0; t < 4; ++t) acc[t] = vzero();

  auto issue_tile = [&](int buf, int k0) {
    if (wid == 0) {
      // A tile: 128 rows x 64 elems, row = 32 DWORDs, pad 4 DWORDs
      tdm_load_2d((u32)(size_t)&sA[buf][0],
                  xb + (size_t)m0blk * 1024 + k0, 64, 128, 1024, 4, 3);
      // B tile: 64 rows x 64 elems
      tdm_load_2d((u32)(size_t)&sB[buf][0],
                  wb + (size_t)n0 * 1024 + k0, 64, 64, 1024, 4, 3);
    }
  };

  issue_tile(0, 0);

  for (int it = 0; it < 16; ++it) {
    if (wid == 0) __builtin_amdgcn_s_wait_tensorcnt(0);
    __syncthreads();
    if (it + 1 < 16) issue_tile((it + 1) & 1, (it + 1) * 64);

    const u16* a0 = &sA[it & 1][(wid * 16 + r15) * SA];
    const u16* b0 = &sB[it & 1][r15 * SB];

    // Batch ALL fragment loads (20 x ds_load_b128 in distinct registers)
    // before the WMMA chain so DS latency overlaps WMMA execution.
    v16bf af0 = load_frag2(a0, kg);
    v16bf af1 = load_frag2(a0 + 32, kg);
    v16bf bf0[4], bf1[4];
#pragma unroll
    for (int t = 0; t < 4; ++t) {
      bf0[t] = load_frag2(b0 + t * 16 * SB, kg);
      bf1[t] = load_frag2(b0 + t * 16 * SB + 32, kg);
    }
#pragma unroll
    for (int t = 0; t < 4; ++t) acc[t] = wmma_bf16(af0, bf0[t], acc[t]);
#pragma unroll
    for (int t = 0; t < 4; ++t) acc[t] = wmma_bf16(af1, bf1[t], acc[t]);
  }

  // Epilogue: scatter to qb/kb [b,h,n,d] and vtb [b,h,d,n] in bf16
  const int m0    = m0blk + wid * 16;
  const int which = n0 >> 10;          // 0=Q, 1=K, 2=V
  const int h     = (n0 & 1023) >> 6;
#pragma unroll
  for (int t = 0; t < 4; ++t) {
#pragma unroll
    for (int r = 0; r < 8; ++r) {
      int m  = m0 + r + 8 * kg;
      int b  = m >> 10;
      int nr = m & 1023;
      int d  = t * 16 + r15;
      float val = acc[t][r];
      size_t bh = (size_t)(b * 16 + h);
      if (which == 0)
        qb[(bh * 1024 + nr) * 64 + d] = f32_bf16(val * ATTN_SCALE);
      else if (which == 1)
        kb[(bh * 1024 + nr) * 64 + d] = f32_bf16(val);
      else
        vtb[(bh * 64 + d) * 1024 + nr] = f32_bf16(val);
    }
  }
}

// ---------------- Kernel 3: flash attention -----------------------
// qb,kb: [b*h, n, 64] bf16 (q pre-scaled); vtb: [b*h, 64, n] bf16.
// Block = 8 waves, 128 q rows per (b,h); K/V^T tiles per 32-wide kv chunk
// shared by all waves, double-buffered via per-lane async LDS loads.
__global__ __launch_bounds__(256) void flash_attn(
    const u16* __restrict__ qb, const u16* __restrict__ kb,
    const u16* __restrict__ vtb, float* __restrict__ out) {
  static constexpr int SK = 72;   // K tile row stride  (32 rows x 64 d)
  static constexpr int SV = 40;   // V^T tile row stride (64 rows x 32 kv)
  __shared__ __align__(16) u16 sK[2][32 * SK];
  __shared__ __align__(16) u16 sV[2][64 * SV];
  __shared__ __align__(16) u16 sP[8][16 * 32];

  const int tid  = threadIdx.x;
  const int lane = tid & 31;
  const int wid  = tid >> 5;
  const int r15  = lane & 15;
  const int kg   = lane >> 4;

  const int bh   = blockIdx.x >> 3;    // 128 (b,h) pairs
  const int qblk = blockIdx.x & 7;
  const int q0   = qblk * 128 + wid * 16;
  const int b    = bh >> 4;
  const int h    = bh & 15;

  // Q A-fragments for this wave's 16 rows (d 0..31 and 32..63)
  const u16* qbase = qb + ((size_t)bh * 1024 + q0 + r15) * 64;
  v16bf qf0 = load_frag2(qbase, kg);
  v16bf qf1 = load_frag2(qbase + 32, kg);

  const u16* kbb  = kb  + (size_t)bh * 1024 * 64;
  const u16* vtbh = vtb + (size_t)bh * 64 * 1024;

  auto issue_kv = [&](int buf, int kv0) {
    // K tile: 32 rows x 64 d = 256 x 16B chunks, 1 per thread
    {
      int row = tid >> 3, ck = tid & 7;
      async_ld_b128(&sK[buf][row * SK + ck * 8],
                    kbb + (size_t)(kv0 + row) * 64 + ck * 8);
    }
    // V^T tile: 64 rows x 32 kv = 256 chunks, 1 per thread
    {
      int row = tid >> 2, ck = tid & 3;
      async_ld_b128(&sV[buf][row * SV + ck * 8],
                    vtbh + (size_t)row * 1024 + kv0 + ck * 8);
    }
  };

  float mrow[8], lrow[8];
  v8f o[4];
#pragma unroll
  for (int t = 0; t < 4; ++t) o[t] = vzero();
#pragma unroll
  for (int r = 0; r < 8; ++r) { mrow[r] = -3.0e38f; lrow[r] = 0.0f; }

  issue_kv(0, 0);

  for (int it = 0; it < 32; ++it) {
    wait_async0();
    __syncthreads();
    if (it + 1 < 32) issue_kv((it + 1) & 1, (it + 1) * 32);

    const u16* kt = &sK[it & 1][0];
    const u16* vt = &sV[it & 1][0];

    // Batch all K fragments (8 x ds_load_b128) and prefetch the V
    // fragments early so their DS latency hides under the softmax VALU.
    v16bf kf0 = load_frag2(kt + r15 * SK, kg);
    v16bf kf1 = load_frag2(kt + r15 * SK + 32, kg);
    v16bf kf2 = load_frag2(kt + (16 + r15) * SK, kg);
    v16bf kf3 = load_frag2(kt + (16 + r15) * SK + 32, kg);
    v16bf vf[4];
#pragma unroll
    for (int t = 0; t < 4; ++t)
      vf[t] = load_frag2(vt + (t * 16 + r15) * SV, kg);

    // S = Q * K^T for this 16x32 kv chunk (two 16x16 C tiles)
    v8f s0 = vzero(), s1 = vzero();
    s0 = wmma_bf16(qf0, kf0, s0);
    s0 = wmma_bf16(qf1, kf1, s0);
    s1 = wmma_bf16(qf0, kf2, s1);
    s1 = wmma_bf16(qf1, kf3, s1);

    // Online softmax (row r + 8*kg lives in 16 lanes of one VGPR)
#pragma unroll
    for (int r = 0; r < 8; ++r) {
      float mx = hmax16(fmaxf(s0[r], s1[r]));
      float mnew  = fmaxf(mrow[r], mx);
      float alpha = __expf(mrow[r] - mnew);
      float p0 = __expf(s0[r] - mnew);
      float p1 = __expf(s1[r] - mnew);
      float rs = hsum16(p0 + p1);
      lrow[r] = lrow[r] * alpha + rs;
      mrow[r] = mnew;
      o[0][r] *= alpha; o[1][r] *= alpha; o[2][r] *= alpha; o[3][r] *= alpha;
      int row = r + 8 * kg;
      sP[wid][row * 32 + r15]      = f32_bf16(p0);
      sP[wid][row * 32 + 16 + r15] = f32_bf16(p1);
    }

    // This wave's cross-lane LDS writes must land before reload
    asm volatile("s_wait_dscnt 0" ::: "memory");

    // P reloaded in A-fragment layout (contraction dim = kv, K=32)
    v16bf pA = load_frag2(&sP[wid][r15 * 32], kg);

    // O += P * V
#pragma unroll
    for (int t = 0; t < 4; ++t)
      o[t] = wmma_bf16(pA, vf[t], o[t]);
  }

  // Normalize and store out[b, n, h*64 + d] (f32)
#pragma unroll
  for (int r = 0; r < 8; ++r) {
    float inv = 1.0f / lrow[r];
    int m = q0 + r + 8 * kg;
    float* orow = out + ((size_t)b * 1024 + m) * 1024 + h * 64;
#pragma unroll
    for (int t = 0; t < 4; ++t)
      orow[t * 16 + r15] = o[t][r] * inv;
  }
}

// ---------------- Host launcher -----------------------------------
extern "C" void kernel_launch(void* const* d_in, const int* in_sizes, int n_in,
                              void* d_out, int out_size, void* d_ws, size_t ws_size,
                              hipStream_t stream) {
  const float* x = (const float*)d_in[0];   // [8,1024,1024]
  const float* w = (const float*)d_in[1];   // [3072,1024]
  float* out = (float*)d_out;               // [8,1024,1024]

  u16* ws = (u16*)d_ws;
  const size_t xn = (size_t)8 * 1024 * 1024;    // x elements
  const size_t wn = (size_t)3072 * 1024;        // w elements
  const size_t qn = (size_t)8 * 16 * 1024 * 64; // per Q/K/V tensor

  u16* xb  = ws;
  u16* wb  = xb + xn;
  u16* qb  = wb + wn;
  u16* kb  = qb + qn;
  u16* vtb = kb + qn;

  cvt_f32_bf16_v8<<<2048, 256, 0, stream>>>(x, xb, (int)(xn / 8));
  cvt_f32_bf16_v8<<<1024, 256, 0, stream>>>(w, wb, (int)(wn / 8));

  // 64 row-blocks (128 rows) x 48 col-blocks (64 cols) = 3072 blocks
  qkv_gemm<<<3072, 256, 0, stream>>>(xb, wb, qb, kb, vtb);

  // 128 (b,h) x 8 q-blocks = 1024 blocks
  flash_attn<<<1024, 256, 0, stream>>>(qb, kb, vtb, out);
}
